// SigmaMoE_88562225643867
// MI455X (gfx1250) — matrix operand order repaired and checked by reading, not compiled, not run on hardware
//
#include <hip/hip_runtime.h>

// ---------------- problem constants (match reference) ----------------
#define T_TOKENS 2048      // B*S
#define DDIM     1024
#define EEXP     8
#define FDIM     1024
#define NPAIRS   (2 * T_TOKENS)          // 4096 (token, expert) pairs
#define PAIR_CAP 4224                    // 4096 + 8*16 alignment padding
#define NTILES   (PAIR_CAP / 16)         // 264 fixed M-tiles
#define NKCH     (DDIM / 32)             // 32 k-chunks (DDIM == FDIM)

typedef __bf16 bf16;
typedef __attribute__((ext_vector_type(4)))  __bf16 v4bf;
typedef __attribute__((ext_vector_type(8)))  __bf16 v8bf;
typedef __attribute__((ext_vector_type(16))) __bf16 v16bf;
typedef __attribute__((ext_vector_type(8)))  float  v8f;
typedef __attribute__((ext_vector_type(4)))  float  v4f;

// Build a 16-element bf16 fragment from two contiguous 16-byte chunks.
__device__ __forceinline__ v16bf load_frag16(const bf16* p0, const bf16* p1) {
  v8bf lo = *(const v8bf*)p0;
  v8bf hi = *(const v8bf*)p1;
  v16bf r;
#pragma unroll
  for (int i = 0; i < 8; ++i) { r[i] = lo[i]; r[i + 8] = hi[i]; }
  return r;
}

// Async global -> LDS 16-byte copy (CDNA5 GLOBAL_LOAD_ASYNC_TO_LDS_B128, GV mode).
// Generic pointers to __shared__ carry the LDS byte offset in addr[31:0].
__device__ __forceinline__ void async_g2l_b128(void* lds_dst, const void* gsrc) {
  unsigned lds = (unsigned)(size_t)lds_dst;
  asm volatile("global_load_async_to_lds_b128 %0, %1, off"
               :: "v"(lds), "v"(gsrc) : "memory");
}
__device__ __forceinline__ void wait_asynccnt0() {
  asm volatile("s_wait_asynccnt 0x0" ::: "memory");
}

// ---------------- init: zero counts, sentinel pair slots ----------------
__global__ void init_kernel(int* counts, int* pair_token, float* pair_gate,
                            int* pair_dst) {
  int i = blockIdx.x * 256 + threadIdx.x;
  if (i < PAIR_CAP) {
    pair_token[i] = 0;
    pair_gate[i]  = 0.0f;
    pair_dst[i]   = -1;
  }
  if (i < EEXP) counts[i] = 0;
}

// ---------------- fp32 -> bf16 conversion (vectorized x4) ----------------
__global__ void convert_kernel(const float* __restrict__ src,
                               bf16* __restrict__ dst, int n4) {
  int i = blockIdx.x * 256 + threadIdx.x;
  if (i < n4) {
    v4f f = *(const v4f*)(src + 4 * (size_t)i);
    v4bf b;
#pragma unroll
    for (int q = 0; q < 4; ++q) b[q] = (bf16)f[q];
    *(v4bf*)(dst + 4 * (size_t)i) = b;
  }
}

// ---------------- routing: logits, top-2, sigmoid gates ----------------
__global__ __launch_bounds__(64) void route_kernel(
    const float* __restrict__ x, const float* __restrict__ esel,
    int* __restrict__ counts, int* __restrict__ idx2, float* __restrict__ g2) {
  __shared__ float sel[8][EEXP];
  const int tid = threadIdx.x;
  const int t8  = tid >> 3;         // token within block (0..7)
  const int e   = tid & 7;          // expert (0..7)
  const int token = blockIdx.x * 8 + t8;

  const float* xr = x + (size_t)token * DDIM;
  const float* er = esel + (size_t)e * DDIM;
  float s = 0.0f;
  for (int i = 0; i < DDIM; i += 4) {
    s += xr[i] * er[i] + xr[i + 1] * er[i + 1] +
         xr[i + 2] * er[i + 2] + xr[i + 3] * er[i + 3];
  }
  sel[t8][e] = s;
  __syncthreads();

  if (e == 0) {
    float b1v = sel[t8][0]; int i1 = 0;
    for (int q = 1; q < EEXP; ++q)
      if (sel[t8][q] > b1v) { b1v = sel[t8][q]; i1 = q; }
    float b2v = -3.4e38f; int i2 = 0;
    for (int q = 0; q < EEXP; ++q)
      if (q != i1 && sel[t8][q] > b2v) { b2v = sel[t8][q]; i2 = q; }
    float ga = 1.0f / (1.0f + __expf(-b1v));
    float gb = 1.0f / (1.0f + __expf(-b2v));
    idx2[2 * token]     = i1;
    idx2[2 * token + 1] = i2;
    g2[2 * token]       = ga;
    g2[2 * token + 1]   = gb;
    atomicAdd(&counts[i1], 1);
    atomicAdd(&counts[i2], 1);
  }
}

// ---------------- scan: 16-aligned segment bases ----------------
__global__ void scan_kernel(const int* __restrict__ counts, int* __restrict__ base,
                            int* __restrict__ cursor) {
  if (threadIdx.x == 0 && blockIdx.x == 0) {
    int acc = 0;
    for (int e = 0; e < EEXP; ++e) {
      base[e]   = acc;
      cursor[e] = acc;
      acc += (counts[e] + 15) & ~15;
    }
    base[EEXP] = acc;
  }
}

// ---------------- scatter pairs into compact per-expert segments ----------------
__global__ __launch_bounds__(64) void scatter_kernel(
    const int* __restrict__ idx2, const float* __restrict__ g2,
    int* __restrict__ cursor, int* __restrict__ pair_token,
    float* __restrict__ pair_gate, int* __restrict__ pair_dst) {
  int i = blockIdx.x * 64 + threadIdx.x;   // pair id = token*2 + kslot
  if (i >= NPAIRS) return;
  int e = idx2[i];
  int slot = atomicAdd(&cursor[e], 1);
  pair_token[slot] = i >> 1;
  pair_gate[slot]  = g2[i];
  pair_dst[slot]   = i;
}

// ---- WMMA building blocks ----
__device__ __forceinline__ void load_bfrags(const bf16* __restrict__ wbase,
                                            int ldk, int n0, int lr, int half,
                                            int kb, v16bf bf[8]) {
#pragma unroll
  for (int j = 0; j < 8; ++j) {
    const bf16* pb = wbase + (size_t)(n0 + j * 16 + lr) * ldk + kb + half * 16;
    bf[j] = load_frag16(pb, pb + 8);
  }
}

__device__ __forceinline__ void wmma8(const bf16* __restrict__ xsrow,
                                      int xs_stride, int kb, int lr, int half,
                                      const v16bf bf[8], v8f acc[8]) {
  const bf16* pa = xsrow + (size_t)lr * xs_stride + kb + half * 16;
  v16bf afrag = load_frag16(pa, pa + 8);
#pragma unroll
  for (int j = 0; j < 8; ++j)
    acc[j] = __builtin_amdgcn_wmma_f32_16x16x32_bf16(
        false, afrag, false, bf[j], (short)0, acc[j], false, false);
}

// Ping-pong double-buffered 16x128-per-wave WMMA mainloop over K (no copies).
__device__ __forceinline__ void wmma_mainloop(const bf16* __restrict__ xsrow,
                                              int xs_stride,
                                              const bf16* __restrict__ wbase,
                                              int ldk, int n0, int lr, int half,
                                              v8f acc[8]) {
  v16bf bufA[8], bufB[8];
  load_bfrags(wbase, ldk, n0, lr, half, 0, bufA);
#pragma unroll 1
  for (int kt = 0; kt < NKCH; kt += 2) {
    load_bfrags(wbase, ldk, n0, lr, half, (kt + 1) * 32, bufB);
    wmma8(xsrow, xs_stride, kt * 32, lr, half, bufA, acc);
    if (kt + 2 < NKCH)
      load_bfrags(wbase, ldk, n0, lr, half, (kt + 2) * 32, bufA);
    wmma8(xsrow, xs_stride, (kt + 1) * 32, lr, half, bufB, acc);
  }
}

// ---------------- grouped GEMM 1: h = relu(x_gathered @ W1e^T + b1e) ----------------
// Block: one 16-row M-tile (16 pairs), 8 waves each owning a 16x128 strip of F.
__global__ __launch_bounds__(256) void gemm1_kernel(
    const bf16* __restrict__ xb, const bf16* __restrict__ W1b,
    const float* __restrict__ b1, const int* __restrict__ base,
    const int* __restrict__ pair_token, bf16* __restrict__ h) {
  __shared__ bf16 xs[16][DDIM + 8];
  __shared__ int  toks[16];

  const int row0 = blockIdx.x * 16;
  if (row0 >= base[EEXP]) return;
  int e = 0;
  while (e < EEXP - 1 && row0 >= base[e + 1]) ++e;

  const int tid  = threadIdx.x;
  const int lane = tid & 31;
  const int lr   = lane & 15;
  const int half = lane >> 4;
  const int n0   = (tid >> 5) * 128;

  const bf16* wbase = W1b + (size_t)e * FDIM * DDIM;

  // warm the first weight k-chunk while x is staged (global_prefetch_b8)
#pragma unroll
  for (int j = 0; j < 8; ++j)
    __builtin_prefetch(wbase + (size_t)(n0 + j * 16 + lr) * DDIM, 0, 0);

  if (tid < 16) {
    int t = pair_token[row0 + tid];
    toks[tid] = (t < 0) ? 0 : t;
  }
  __syncthreads();

  {  // stage gathered x rows -> LDS via async global->LDS copies
    const int row = tid >> 4;
    const int c0  = tid & 15;
    const bf16* src = xb + (size_t)toks[row] * DDIM;
#pragma unroll
    for (int j = 0; j < 8; ++j) {
      int col = (c0 + 16 * j) * 8;
      async_g2l_b128(&xs[row][col], &src[col]);
    }
    wait_asynccnt0();
  }
  __syncthreads();

  v8f acc[8];
#pragma unroll
  for (int j = 0; j < 8; ++j)
#pragma unroll
    for (int q = 0; q < 8; ++q) acc[j][q] = 0.0f;

  wmma_mainloop(&xs[0][0], DDIM + 8, wbase, DDIM, n0, lr, half, acc);

  // epilogue: + bias, relu, store bf16 h[slot][F]
#pragma unroll
  for (int j = 0; j < 8; ++j) {
    int n = n0 + j * 16 + lr;
    float bias = b1[e * FDIM + n];
#pragma unroll
    for (int v = 0; v < 8; ++v) {
      int m = v + half * 8;
      float val = acc[j][v] + bias;
      val = val > 0.0f ? val : 0.0f;
      h[(size_t)(row0 + m) * FDIM + n] = (bf16)val;
    }
  }
}

// ---------------- grouped GEMM 2: y[dst] = gate * (h @ W2e^T) ----------------
__global__ __launch_bounds__(256) void gemm2_kernel(
    const bf16* __restrict__ h, const bf16* __restrict__ W2b,
    const int* __restrict__ base, const float* __restrict__ pair_gate,
    const int* __restrict__ pair_dst, float* __restrict__ y) {
  __shared__ bf16  hs[16][FDIM + 8];
  __shared__ float gates[16];
  __shared__ int   dsts[16];

  const int row0 = blockIdx.x * 16;
  if (row0 >= base[EEXP]) return;
  int e = 0;
  while (e < EEXP - 1 && row0 >= base[e + 1]) ++e;

  const int tid  = threadIdx.x;
  const int lane = tid & 31;
  const int lr   = lane & 15;
  const int half = lane >> 4;
  const int n0   = (tid >> 5) * 128;

  const bf16* wbase = W2b + (size_t)e * DDIM * FDIM;

#pragma unroll
  for (int j = 0; j < 8; ++j)
    __builtin_prefetch(wbase + (size_t)(n0 + j * 16 + lr) * FDIM, 0, 0);

  if (tid < 16) {
    gates[tid] = pair_gate[row0 + tid];
    dsts[tid]  = pair_dst[row0 + tid];
  }

  {  // stage h rows -> LDS via async global->LDS copies
    const int row = tid >> 4;
    const int c0  = tid & 15;
    const bf16* src = h + (size_t)(row0 + row) * FDIM;
#pragma unroll
    for (int j = 0; j < 8; ++j) {
      int col = (c0 + 16 * j) * 8;
      async_g2l_b128(&hs[row][col], &src[col]);
    }
    wait_asynccnt0();
  }
  __syncthreads();

  v8f acc[8];
#pragma unroll
  for (int j = 0; j < 8; ++j)
#pragma unroll
    for (int q = 0; q < 8; ++q) acc[j][q] = 0.0f;

  wmma_mainloop(&hs[0][0], FDIM + 8, wbase, FDIM, n0, lr, half, acc);

#pragma unroll
  for (int j = 0; j < 8; ++j) {
    int n = n0 + j * 16 + lr;
#pragma unroll
    for (int v = 0; v < 8; ++v) {
      int m   = v + half * 8;
      int dst = dsts[m];
      if (dst >= 0) y[(size_t)dst * DDIM + n] = gates[m] * acc[j][v];
    }
  }
}

// ---------------- finalize: combine 2 expert paths + gated b2 + loss scalar ----------------
__global__ void finalize_kernel(const float* __restrict__ y,
                                const float* __restrict__ b2,
                                const int* __restrict__ idx2,
                                const float* __restrict__ g2,
                                float* __restrict__ out) {
  int gid = blockIdx.x * 256 + threadIdx.x;
  if (gid >= T_TOKENS * DDIM) return;
  int t = gid >> 10;
  int d = gid & 1023;
  int e0 = idx2[2 * t], e1 = idx2[2 * t + 1];
  float r = y[(size_t)(2 * t) * DDIM + d] + y[(size_t)(2 * t + 1) * DDIM + d] +
            g2[2 * t] * b2[e0 * DDIM + d] + g2[2 * t + 1] * b2[e1 * DDIM + d];
  out[gid] = r;
  if (gid == 0) out[T_TOKENS * DDIM] = 0.0f;  // reg loss scalar (eval mode: 0)
}

// ---------------- host launcher ----------------
extern "C" void kernel_launch(void* const* d_in, const int* in_sizes, int n_in,
                              void* d_out, int out_size, void* d_ws, size_t ws_size,
                              hipStream_t stream) {
  const float* x    = (const float*)d_in[0];
  const float* esel = (const float*)d_in[1];
  const float* W1   = (const float*)d_in[2];
  const float* b1   = (const float*)d_in[3];
  const float* W2   = (const float*)d_in[4];
  const float* b2   = (const float*)d_in[5];
  float* out = (float*)d_out;

  char* ws = (char*)d_ws;
  size_t off = 0;
  auto alloc = [&](size_t bytes) -> void* {
    void* p = ws + off;
    off = (off + bytes + 255) & ~(size_t)255;
    return p;
  };

  int*   counts     = (int*)  alloc(EEXP * sizeof(int));
  int*   base       = (int*)  alloc((EEXP + 1) * sizeof(int));
  int*   cursor     = (int*)  alloc(EEXP * sizeof(int));
  int*   idx2       = (int*)  alloc(NPAIRS * sizeof(int));
  float* g2         = (float*)alloc(NPAIRS * sizeof(float));
  int*   pair_token = (int*)  alloc(PAIR_CAP * sizeof(int));
  float* pair_gate  = (float*)alloc(PAIR_CAP * sizeof(float));
  int*   pair_dst   = (int*)  alloc(PAIR_CAP * sizeof(int));
  bf16*  xb         = (bf16*) alloc((size_t)T_TOKENS * DDIM * sizeof(bf16));
  bf16*  W1b        = (bf16*) alloc((size_t)EEXP * FDIM * DDIM * sizeof(bf16));
  bf16*  W2b        = (bf16*) alloc((size_t)EEXP * DDIM * FDIM * sizeof(bf16));
  bf16*  h          = (bf16*) alloc((size_t)PAIR_CAP * FDIM * sizeof(bf16));
  float* y          = (float*)alloc((size_t)NPAIRS * DDIM * sizeof(float));

  init_kernel<<<(PAIR_CAP + 255) / 256, 256, 0, stream>>>(counts, pair_token,
                                                          pair_gate, pair_dst);

  int nx4 = T_TOKENS * DDIM / 4;
  int nw4 = EEXP * FDIM * DDIM / 4;
  convert_kernel<<<(nx4 + 255) / 256, 256, 0, stream>>>(x, xb, nx4);
  convert_kernel<<<(nw4 + 255) / 256, 256, 0, stream>>>(W1, W1b, nw4);
  convert_kernel<<<(nw4 + 255) / 256, 256, 0, stream>>>(W2, W2b, nw4);

  route_kernel<<<T_TOKENS / 8, 64, 0, stream>>>(x, esel, counts, idx2, g2);
  scan_kernel<<<1, 1, 0, stream>>>(counts, base, cursor);
  scatter_kernel<<<NPAIRS / 64, 64, 0, stream>>>(idx2, g2, cursor, pair_token,
                                                 pair_gate, pair_dst);

  gemm1_kernel<<<NTILES, 256, 0, stream>>>(xb, W1b, b1, base, pair_token, h);
  gemm2_kernel<<<NTILES, 256, 0, stream>>>(h, W2b, base, pair_gate, pair_dst, y);

  finalize_kernel<<<(T_TOKENS * DDIM + 255) / 256, 256, 0, stream>>>(y, b2, idx2,
                                                                    g2, out);
}
